// FullPairformerBlock_7164005450352
// MI455X (gfx1250) — matrix operand order repaired and checked by reading, not compiled, not run on hardware
//
#include <hip/hip_runtime.h>

// ---------------------------------------------------------------------------
// Types / WMMA helpers (gfx1250, wave32)
// ---------------------------------------------------------------------------
typedef __bf16 v16bf __attribute__((ext_vector_type(16)));
typedef float  v8f   __attribute__((ext_vector_type(8)));
typedef unsigned int uint4v __attribute__((ext_vector_type(4)));

// A operand: 16x32 bf16 tile, row-major source with leading dim `ld` (elements).
// Lane L<16 : row L,   K = {0..7, 16..23}
// Lane L>=16: row L-16,K = {8..15, 24..31}
__device__ __forceinline__ v16bf load_a16x32(const __bf16* p, int ld, int lane) {
  const __bf16* base = p + (size_t)(lane & 15) * ld + ((lane >> 4) << 3);
  v16bf out;
  ((uint4v*)&out)[0] = *(const uint4v*)(base);
  ((uint4v*)&out)[1] = *(const uint4v*)(base + 16);
  return out;
}

// B operand: 32x16 bf16 tile. Source is W[N][K] row-major (i.e. B^T), so
// lane L holds column n=L&15, K = {0..7,16..23} (L<16) or {8..15,24..31} (L>=16).
__device__ __forceinline__ v16bf load_b32x16(const __bf16* p, int ld, int lane) {
  const __bf16* base = p + (size_t)(lane & 15) * ld + ((lane >> 4) << 3);
  v16bf out;
  ((uint4v*)&out)[0] = *(const uint4v*)(base);
  ((uint4v*)&out)[1] = *(const uint4v*)(base + 16);
  return out;
}

__device__ __forceinline__ v8f wmma_bf16(v16bf a, v16bf b, v8f c) {
  return __builtin_amdgcn_wmma_f32_16x16x32_bf16(false, a, false, b, (short)0, c,
                                                 false, false);
}

__device__ __forceinline__ float sigmf(float x) { return 1.f / (1.f + __expf(-x)); }

// ---------------------------------------------------------------------------
// Weight convert+transpose: src fp32 [K,N] -> dst bf16 [Npad][K] (zero padded)
// ---------------------------------------------------------------------------
__global__ __launch_bounds__(256) void wconv_kernel(const float* __restrict__ src,
                                                    __bf16* __restrict__ dst,
                                                    int K, int N, int Npad) {
  int idx = blockIdx.x * 256 + threadIdx.x;
  if (idx >= Npad * K) return;
  int n = idx / K, k = idx - n * K;
  float v = (n < N) ? src[(size_t)k * N + n] : 0.f;
  dst[idx] = (__bf16)v;
}

// ---------------------------------------------------------------------------
// LayerNorm over last dim -> bf16. One wave per row. transposeRead swaps the
// two 256-wide spatial indices of the token (used by the "ending" attention).
// ---------------------------------------------------------------------------
__global__ __launch_bounds__(256) void ln_kernel(const float* __restrict__ X,
                                                 const float* __restrict__ gw,
                                                 const float* __restrict__ bw,
                                                 __bf16* __restrict__ out,
                                                 int M, int C, int transposeRead) {
  int row = blockIdx.x * 8 + (threadIdx.x >> 5);
  if (row >= M) return;
  int lane = threadIdx.x & 31;
  int src = transposeRead ? ((row & 255) * 256 + (row >> 8)) : row;
  const float* x = X + (size_t)src * C;
  float s = 0.f, s2 = 0.f;
  for (int c = lane; c < C; c += 32) { float v = x[c]; s += v; s2 += v * v; }
#pragma unroll
  for (int m = 16; m >= 1; m >>= 1) {
    s  += __shfl_xor(s,  m, 32);
    s2 += __shfl_xor(s2, m, 32);
  }
  float mean = s / C;
  float var  = s2 / C - mean * mean;
  float inv  = rsqrtf(var + 1e-5f);
  __bf16* o = out + (size_t)row * C;
  for (int c = lane; c < C; c += 32)
    o[c] = (__bf16)((x[c] - mean) * inv * gw[c] + bw[c]);
}

// ---------------------------------------------------------------------------
// Generic bf16 WMMA GEMM:  acc1 = A1 @ W1^T, acc2 = A2 @ W2^T (dual EPIs).
// A: [M,K] bf16 row-major.  W: [Npad,K] bf16 row-major (pre-transposed).
// Block 256 = 8 waves; block tile 128(M) x 64(N); wave tile 16 x 64.
//
// EPI: 0 store f32        1 f32 +=          2 store bf16 (+bias vec)
//      3 sigmoid(acc1)*acc2 -> bf16         4 silu(acc1)*acc2 -> bf16
//      5 f32 += sigmoid(acc1)*acc2          6 sigmoid(acc1) -> bf16
// storeMode: 0 row*ld+col   1 col*M+row (channel-major)
//            2 col*M + (row&255)*256 + row>>8   (channel-major, transposed)
//            3 ((row&255)*256 + row>>8)*ld + col (token-transposed residual)
//            4 (col>>5)*sH + (row>>8)*sI + (row&255)*32 + (col&31)  (Q/K/G)
//            5 (col>>5)*sH + (row>>8)*sI + (col&31)*256 + (row&255) (V^T)
// ---------------------------------------------------------------------------
template <int EPI>
__global__ __launch_bounds__(256) void gemm_kernel(
    const __bf16* __restrict__ A1, const __bf16* __restrict__ W1,
    const __bf16* __restrict__ A2, const __bf16* __restrict__ W2,
    const float* __restrict__ biasVec, float* __restrict__ outF,
    __bf16* __restrict__ outB, int M, int Npad, int Nstore, int K,
    int storeMode, int sH, int sI) {
  constexpr bool DUAL = (EPI == 3 || EPI == 4 || EPI == 5);
  int wave = threadIdx.x >> 5, lane = threadIdx.x & 31;
  int m0 = blockIdx.y * 128 + wave * 16;
  int n0 = blockIdx.x * 64;

  v8f acc1[4], acc2[4];
#pragma unroll
  for (int i = 0; i < 4; ++i)
#pragma unroll
    for (int j = 0; j < 8; ++j) { acc1[i][j] = 0.f; acc2[i][j] = 0.f; }

  for (int k = 0; k < K; k += 32) {
    v16bf a1 = load_a16x32(A1 + (size_t)m0 * K + k, K, lane);
    v16bf a2;
    if constexpr (DUAL) a2 = load_a16x32(A2 + (size_t)m0 * K + k, K, lane);
#pragma unroll
    for (int nt = 0; nt < 4; ++nt) {
      int nn = n0 + nt * 16;
      if (nn < Npad) {
        v16bf b1 = load_b32x16(W1 + (size_t)nn * K + k, K, lane);
        acc1[nt] = wmma_bf16(a1, b1, acc1[nt]);
        if constexpr (DUAL) {
          v16bf b2 = load_b32x16(W2 + (size_t)nn * K + k, K, lane);
          acc2[nt] = wmma_bf16(a2, b2, acc2[nt]);
        }
      }
    }
  }

#pragma unroll
  for (int nt = 0; nt < 4; ++nt) {
    int nn = n0 + nt * 16;
    if (nn >= Npad) continue;
#pragma unroll
    for (int r = 0; r < 8; ++r) {
      int col = nn + (lane & 15);
      int row = m0 + r + ((lane >> 4) << 3);
      if (col >= Nstore || row >= M) continue;
      float x1 = acc1[nt][r];
      float x2 = acc2[nt][r];
      if (biasVec) x1 += biasVec[col];
      float val;
      if constexpr (EPI == 0 || EPI == 1 || EPI == 2) val = x1;
      else if constexpr (EPI == 3 || EPI == 5) val = sigmf(x1) * x2;
      else if constexpr (EPI == 4) val = x1 * sigmf(x1) * x2;
      else val = sigmf(x1);
      size_t idx;
      if (storeMode == 0)      idx = (size_t)row * Nstore + col;
      else if (storeMode == 1) idx = (size_t)col * M + row;
      else if (storeMode == 2) idx = (size_t)col * M + (size_t)(row & 255) * 256 + (row >> 8);
      else if (storeMode == 3) idx = ((size_t)(row & 255) * 256 + (row >> 8)) * (size_t)Nstore + col;
      else if (storeMode == 4) idx = (size_t)(col >> 5) * sH + (size_t)(row >> 8) * sI +
                                     (size_t)(row & 255) * 32 + (col & 31);
      else                     idx = (size_t)(col >> 5) * sH + (size_t)(row >> 8) * sI +
                                     (size_t)(col & 31) * 256 + (row & 255);
      if constexpr (EPI == 1 || EPI == 5) outF[idx] += val;
      else if constexpr (EPI == 0)        outF[idx] = val;
      else                                outB[idx] = (__bf16)val;
    }
  }
}

// ---------------------------------------------------------------------------
// Triangle einsum: per channel c, T_c = A_c(256x256) @ B_c(256x256)^T,
// output scattered to [i][j][c] fp32. grid = (4 jtiles, 2 itiles, 128 ch)
// ---------------------------------------------------------------------------
__global__ __launch_bounds__(256) void tri_einsum_kernel(
    const __bf16* __restrict__ A, const __bf16* __restrict__ B,
    float* __restrict__ T) {
  int c = blockIdx.z;
  int wave = threadIdx.x >> 5, lane = threadIdx.x & 31;
  const __bf16* Ac = A + (size_t)c * 65536;
  const __bf16* Bc = B + (size_t)c * 65536;
  int m0 = blockIdx.y * 128 + wave * 16;
  int n0 = blockIdx.x * 64;
  v8f acc[4];
#pragma unroll
  for (int i = 0; i < 4; ++i)
#pragma unroll
    for (int j = 0; j < 8; ++j) acc[i][j] = 0.f;
  for (int k = 0; k < 256; k += 32) {
    v16bf a = load_a16x32(Ac + (size_t)m0 * 256 + k, 256, lane);
#pragma unroll
    for (int nt = 0; nt < 4; ++nt) {
      v16bf b = load_b32x16(Bc + (size_t)(n0 + nt * 16) * 256 + k, 256, lane);
      acc[nt] = wmma_bf16(a, b, acc[nt]);
    }
  }
#pragma unroll
  for (int nt = 0; nt < 4; ++nt) {
    int col = n0 + nt * 16 + (lane & 15);
#pragma unroll
    for (int r = 0; r < 8; ++r) {
      int row = m0 + r + ((lane >> 4) << 3);
      T[((size_t)row * 256 + col) * 128 + c] = acc[nt][r];
    }
  }
}

// ---------------------------------------------------------------------------
// Attention logits + softmax. One block (128 thr = 4 waves) per instance g.
// Q,K layouts: instance base g*8192, row j stride 32, d contiguous.
// bias slab: bias + (g>>instShift)*65536, [jq][k].  P out: [g][jq][k] bf16.
// ---------------------------------------------------------------------------
__global__ __launch_bounds__(128) void attn_softmax_kernel(
    const __bf16* __restrict__ Q, const __bf16* __restrict__ Km,
    const float* __restrict__ bias, __bf16* __restrict__ P, int instShift,
    float scale) {
  __shared__ float smem[4 * 16 * 256];
  int g = blockIdx.x;
  int wave = threadIdx.x >> 5, lane = threadIdx.x & 31;
  const __bf16* Qg = Q + (size_t)g * 8192;
  const __bf16* Kg = Km + (size_t)g * 8192;
  const float* Bs = bias + (size_t)(g >> instShift) * 65536;
  __bf16* Pg = P + (size_t)g * 65536;
  float* S = smem + wave * (16 * 256);
  for (int qt = 0; qt < 4; ++qt) {
    int q0 = wave * 64 + qt * 16;
    v16bf a = load_a16x32(Qg + (size_t)q0 * 32, 32, lane);
#pragma unroll
    for (int kt = 0; kt < 16; ++kt) {
      v8f acc;
#pragma unroll
      for (int j = 0; j < 8; ++j) acc[j] = 0.f;
      v16bf b = load_b32x16(Kg + (size_t)kt * 16 * 32, 32, lane);
      acc = wmma_bf16(a, b, acc);
      int cc = kt * 16 + (lane & 15);
      int rb = (lane >> 4) << 3;
#pragma unroll
      for (int r = 0; r < 8; ++r)
        S[(rb + r) * 256 + cc] = acc[r] * scale + Bs[(size_t)(q0 + rb + r) * 256 + cc];
    }
    __syncthreads();
    int rw = lane & 15, half = lane >> 4;
    float* Srow = S + rw * 256 + half * 128;
    float mx = -3.0e38f;
    for (int c = 0; c < 128; ++c) mx = fmaxf(mx, Srow[c]);
    mx = fmaxf(mx, __shfl_xor(mx, 16, 32));
    float sum = 0.f;
    for (int c = 0; c < 128; ++c) { float e = __expf(Srow[c] - mx); Srow[c] = e; sum += e; }
    sum += __shfl_xor(sum, 16, 32);
    float inv = 1.f / sum;
    __bf16* prow = Pg + (size_t)(q0 + rw) * 256 + half * 128;
    for (int c = 0; c < 128; c += 8) {
      __bf16 tmp[8];
#pragma unroll
      for (int u = 0; u < 8; ++u) tmp[u] = (__bf16)(Srow[c + u] * inv);
      *(uint4v*)(prow + c) = *(const uint4v*)tmp;
    }
    __syncthreads();
  }
}

// ---------------------------------------------------------------------------
// O = P @ V  (V stored transposed [d][k], instance base g*8192), gated by
// pre-sigmoided G (layout like Q), written to go[t][h*32+d] bf16.
// t = (g & ((1<<instShift)-1))*256 + jq.
// ---------------------------------------------------------------------------
__global__ __launch_bounds__(256) void attn_pv_kernel(
    const __bf16* __restrict__ P, const __bf16* __restrict__ Vt,
    const __bf16* __restrict__ G, __bf16* __restrict__ go, int instShift,
    int HD) {
  int g = blockIdx.x;
  int wave = threadIdx.x >> 5, lane = threadIdx.x & 31;
  int h = g >> instShift;
  int iPart = g - (h << instShift);
  const __bf16* Pg = P + (size_t)g * 65536;
  const __bf16* Vg = Vt + (size_t)g * 8192;
  const __bf16* Gg = G + (size_t)g * 8192;
  for (int qt = 0; qt < 2; ++qt) {
    int q0 = wave * 32 + qt * 16;
    v8f acc[2];
#pragma unroll
    for (int i = 0; i < 2; ++i)
#pragma unroll
      for (int j = 0; j < 8; ++j) acc[i][j] = 0.f;
    for (int k = 0; k < 256; k += 32) {
      v16bf a = load_a16x32(Pg + (size_t)q0 * 256 + k, 256, lane);
#pragma unroll
      for (int nt = 0; nt < 2; ++nt) {
        v16bf b = load_b32x16(Vg + (size_t)nt * 16 * 256 + k, 256, lane);
        acc[nt] = wmma_bf16(a, b, acc[nt]);
      }
    }
#pragma unroll
    for (int nt = 0; nt < 2; ++nt) {
      int d = nt * 16 + (lane & 15);
#pragma unroll
      for (int r = 0; r < 8; ++r) {
        int jq = q0 + r + ((lane >> 4) << 3);
        float gate = (float)Gg[(size_t)jq * 32 + d];
        float val = acc[nt][r] * gate;
        size_t t = (size_t)iPart * 256 + jq;
        go[t * HD + h * 32 + d] = (__bf16)val;
      }
    }
  }
}

// ---------------------------------------------------------------------------
// Host-side dispatch
// ---------------------------------------------------------------------------
static void launch_gemm(hipStream_t st, int EPI, const __bf16* A1,
                        const __bf16* W1, const __bf16* A2, const __bf16* W2,
                        const float* biasVec, float* outF, __bf16* outB, int M,
                        int Npad, int Nstore, int K, int mode, int sH, int sI) {
  dim3 grid((Npad + 63) / 64, (M + 127) / 128);
  dim3 block(256);
  switch (EPI) {
    case 0: gemm_kernel<0><<<grid, block, 0, st>>>(A1, W1, A2, W2, biasVec, outF, outB, M, Npad, Nstore, K, mode, sH, sI); break;
    case 1: gemm_kernel<1><<<grid, block, 0, st>>>(A1, W1, A2, W2, biasVec, outF, outB, M, Npad, Nstore, K, mode, sH, sI); break;
    case 2: gemm_kernel<2><<<grid, block, 0, st>>>(A1, W1, A2, W2, biasVec, outF, outB, M, Npad, Nstore, K, mode, sH, sI); break;
    case 3: gemm_kernel<3><<<grid, block, 0, st>>>(A1, W1, A2, W2, biasVec, outF, outB, M, Npad, Nstore, K, mode, sH, sI); break;
    case 4: gemm_kernel<4><<<grid, block, 0, st>>>(A1, W1, A2, W2, biasVec, outF, outB, M, Npad, Nstore, K, mode, sH, sI); break;
    case 5: gemm_kernel<5><<<grid, block, 0, st>>>(A1, W1, A2, W2, biasVec, outF, outB, M, Npad, Nstore, K, mode, sH, sI); break;
    default: gemm_kernel<6><<<grid, block, 0, st>>>(A1, W1, A2, W2, biasVec, outF, outB, M, Npad, Nstore, K, mode, sH, sI); break;
  }
}

extern "C" void kernel_launch(void* const* d_in, const int* in_sizes, int n_in,
                              void* d_out, int out_size, void* d_ws,
                              size_t ws_size, hipStream_t stream) {
  (void)in_sizes; (void)n_in; (void)out_size; (void)ws_size;
  const float SCALE_ATT = 0.17677669529663687f;  // 1/sqrt(32)

  float* Sw = (float*)d_out;           // [256,384]
  float* Zw = (float*)d_out + 98304;   // [256,256,128]
  hipMemcpyAsync(Sw, d_in[0], (size_t)98304 * 4, hipMemcpyDeviceToDevice, stream);
  hipMemcpyAsync(Zw, d_in[1], (size_t)8388608 * 4, hipMemcpyDeviceToDevice, stream);

  // ---- workspace carve (deterministic) ----
  char* ws = (char*)d_ws;
  size_t off = 0;
  auto takeB = [&](size_t elems) { __bf16* p = (__bf16*)(ws + off); off = (off + elems * 2 + 255) & ~(size_t)255; return p; };
  auto takeF = [&](size_t elems) { float* p = (float*)(ws + off); off = (off + elems * 4 + 255) & ~(size_t)255; return p; };

  __bf16* znB  = takeB(8388608);   // LN'd activations (bf16)
  __bf16* bufA = takeB(8388608);   // a / Q / tn
  __bf16* bufB = takeB(8388608);   // b / K
  __bf16* bufV = takeB(8388608);   // V^T
  __bf16* bufG = takeB(8388608);   // sigmoid gates
  __bf16* goB  = takeB(8388608);   // gated attention output
  __bf16* snB  = takeB(98304);     // LN'd S
  float*  bufT  = takeF(8388608);  // triangle product (fp32, pre-LN2)
  float*  biasF = takeF(16 * 65536);
  __bf16* bufP = takeB(67108864);  // softmax probs / transition hidden

  auto wconv = [&](int pidx, __bf16* dst, int K, int N, int Npad) {
    int tot = Npad * K;
    wconv_kernel<<<(tot + 255) / 256, 256, 0, stream>>>((const float*)d_in[pidx], dst, K, N, Npad);
  };
  auto ln = [&](const float* X, int gi, int bi, __bf16* out, int M, int C, int tr) {
    ln_kernel<<<(M + 7) / 8, 256, 0, stream>>>(X, (const float*)d_in[gi], (const float*)d_in[bi], out, M, C, tr);
  };

  // ---- convert + transpose all weights to bf16 [N][K] ----
  // tmo: 2 ln.g 3 ln.b 4 w_a 5 wg_a 6 w_b 7 wg_b 8 ln2.g 9 ln2.b 10 w_o 11 wg_o
  __bf16 *tmo_wa = takeB(16384), *tmo_wga = takeB(16384), *tmo_wb = takeB(16384),
         *tmo_wgb = takeB(16384), *tmo_wo = takeB(16384), *tmo_wgo = takeB(16384);
  wconv(4, tmo_wa, 128, 128, 128); wconv(5, tmo_wga, 128, 128, 128);
  wconv(6, tmo_wb, 128, 128, 128); wconv(7, tmo_wgb, 128, 128, 128);
  wconv(10, tmo_wo, 128, 128, 128); wconv(11, tmo_wgo, 128, 128, 128);
  __bf16 *tmi_wa = takeB(16384), *tmi_wga = takeB(16384), *tmi_wb = takeB(16384),
         *tmi_wgb = takeB(16384), *tmi_wo = takeB(16384), *tmi_wgo = takeB(16384);
  wconv(14, tmi_wa, 128, 128, 128); wconv(15, tmi_wga, 128, 128, 128);
  wconv(16, tmi_wb, 128, 128, 128); wconv(17, tmi_wgb, 128, 128, 128);
  wconv(20, tmi_wo, 128, 128, 128); wconv(21, tmi_wgo, 128, 128, 128);
  // tas: 22 ln.g 23 ln.b 24 wq 25 wk 26 wv 27 wb 28 wg 29 wo
  __bf16 *tas_wq = takeB(16384), *tas_wk = takeB(16384), *tas_wv = takeB(16384),
         *tas_wg = takeB(16384), *tas_wo = takeB(16384), *tas_wbz = takeB(2048);
  wconv(24, tas_wq, 128, 128, 128); wconv(25, tas_wk, 128, 128, 128);
  wconv(26, tas_wv, 128, 128, 128); wconv(28, tas_wg, 128, 128, 128);
  wconv(29, tas_wo, 128, 128, 128); wconv(27, tas_wbz, 128, 4, 16);
  __bf16 *tae_wq = takeB(16384), *tae_wk = takeB(16384), *tae_wv = takeB(16384),
         *tae_wg = takeB(16384), *tae_wo = takeB(16384), *tae_wbz = takeB(2048);
  wconv(32, tae_wq, 128, 128, 128); wconv(33, tae_wk, 128, 128, 128);
  wconv(34, tae_wv, 128, 128, 128); wconv(36, tae_wg, 128, 128, 128);
  wconv(37, tae_wo, 128, 128, 128); wconv(35, tae_wbz, 128, 4, 16);
  // zt: 38 ln.g 39 ln.b 40 w1 41 w2 42 w3
  __bf16 *zt_w1 = takeB(65536), *zt_w2 = takeB(65536), *zt_w3 = takeB(65536);
  wconv(40, zt_w1, 128, 512, 512); wconv(41, zt_w2, 128, 512, 512);
  wconv(42, zt_w3, 512, 128, 128);
  // apb: 43 ln_s.g 44 ln_s.b 45 wq 46 bq 47 wk 48 wv 49 wg 50 ln_z.g 51 ln_z.b 52 wb 53 wo
  __bf16 *apb_wq = takeB(147456), *apb_wk = takeB(147456), *apb_wv = takeB(147456),
         *apb_wg = takeB(147456), *apb_wo = takeB(147456), *apb_wbz = takeB(2048);
  wconv(45, apb_wq, 384, 384, 384); wconv(47, apb_wk, 384, 384, 384);
  wconv(48, apb_wv, 384, 384, 384); wconv(49, apb_wg, 384, 384, 384);
  wconv(53, apb_wo, 384, 384, 384); wconv(52, apb_wbz, 128, 12, 16);
  // st: 54 ln.g 55 ln.b 56 w1 57 w2 58 w3
  __bf16 *st_w1 = takeB(589824), *st_w2 = takeB(589824), *st_w3 = takeB(589824);
  wconv(56, st_w1, 384, 1536, 1536); wconv(57, st_w2, 384, 1536, 1536);
  wconv(58, st_w3, 1536, 384, 384);

  // =================== triangle mult (outgoing) ===================
  ln(Zw, 2, 3, znB, 65536, 128, 0);
  launch_gemm(stream, 3, znB, tmo_wga, znB, tmo_wa, nullptr, nullptr, bufA, 65536, 128, 128, 128, 1, 0, 0);
  launch_gemm(stream, 3, znB, tmo_wgb, znB, tmo_wb, nullptr, nullptr, bufB, 65536, 128, 128, 128, 1, 0, 0);
  tri_einsum_kernel<<<dim3(4, 2, 128), 256, 0, stream>>>(bufA, bufB, bufT);
  ln(bufT, 8, 9, bufA, 65536, 128, 0);  // ln2 -> tn (reuse bufA)
  launch_gemm(stream, 5, znB, tmo_wgo, bufA, tmo_wo, nullptr, Zw, nullptr, 65536, 128, 128, 128, 0, 0, 0);

  // =================== triangle mult (incoming) ===================
  ln(Zw, 12, 13, znB, 65536, 128, 0);
  launch_gemm(stream, 3, znB, tmi_wga, znB, tmi_wa, nullptr, nullptr, bufA, 65536, 128, 128, 128, 2, 0, 0);
  launch_gemm(stream, 3, znB, tmi_wgb, znB, tmi_wb, nullptr, nullptr, bufB, 65536, 128, 128, 128, 2, 0, 0);
  tri_einsum_kernel<<<dim3(4, 2, 128), 256, 0, stream>>>(bufA, bufB, bufT);
  ln(bufT, 18, 19, bufA, 65536, 128, 0);
  launch_gemm(stream, 5, znB, tmi_wgo, bufA, tmi_wo, nullptr, Zw, nullptr, 65536, 128, 128, 128, 0, 0, 0);

  // =================== triangle attention (starting) ===================
  ln(Zw, 22, 23, znB, 65536, 128, 0);
  launch_gemm(stream, 2, znB, tas_wq, nullptr, nullptr, nullptr, nullptr, bufA, 65536, 128, 128, 128, 4, 2097152, 8192);
  launch_gemm(stream, 2, znB, tas_wk, nullptr, nullptr, nullptr, nullptr, bufB, 65536, 128, 128, 128, 4, 2097152, 8192);
  launch_gemm(stream, 2, znB, tas_wv, nullptr, nullptr, nullptr, nullptr, bufV, 65536, 128, 128, 128, 5, 2097152, 8192);
  launch_gemm(stream, 6, znB, tas_wg, nullptr, nullptr, nullptr, nullptr, bufG, 65536, 128, 128, 128, 4, 2097152, 8192);
  launch_gemm(stream, 0, znB, tas_wbz, nullptr, nullptr, nullptr, biasF, nullptr, 65536, 16, 4, 128, 1, 0, 0);
  attn_softmax_kernel<<<1024, 128, 0, stream>>>(bufA, bufB, biasF, bufP, 8, SCALE_ATT);
  attn_pv_kernel<<<1024, 256, 0, stream>>>(bufP, bufV, bufG, goB, 8, 128);
  launch_gemm(stream, 1, goB, tas_wo, nullptr, nullptr, nullptr, Zw, nullptr, 65536, 128, 128, 128, 0, 0, 0);

  // =================== triangle attention (ending) ===================
  ln(Zw, 30, 31, znB, 65536, 128, 1);  // transposed read
  launch_gemm(stream, 2, znB, tae_wq, nullptr, nullptr, nullptr, nullptr, bufA, 65536, 128, 128, 128, 4, 2097152, 8192);
  launch_gemm(stream, 2, znB, tae_wk, nullptr, nullptr, nullptr, nullptr, bufB, 65536, 128, 128, 128, 4, 2097152, 8192);
  launch_gemm(stream, 2, znB, tae_wv, nullptr, nullptr, nullptr, nullptr, bufV, 65536, 128, 128, 128, 5, 2097152, 8192);
  launch_gemm(stream, 6, znB, tae_wg, nullptr, nullptr, nullptr, nullptr, bufG, 65536, 128, 128, 128, 4, 2097152, 8192);
  launch_gemm(stream, 0, znB, tae_wbz, nullptr, nullptr, nullptr, biasF, nullptr, 65536, 16, 4, 128, 1, 0, 0);
  attn_softmax_kernel<<<1024, 128, 0, stream>>>(bufA, bufB, biasF, bufP, 8, SCALE_ATT);
  attn_pv_kernel<<<1024, 256, 0, stream>>>(bufP, bufV, bufG, goB, 8, 128);
  launch_gemm(stream, 1, goB, tae_wo, nullptr, nullptr, nullptr, Zw, nullptr, 65536, 128, 128, 128, 3, 0, 0);  // transposed residual

  // =================== Z transition ===================
  ln(Zw, 38, 39, znB, 65536, 128, 0);
  launch_gemm(stream, 4, znB, zt_w1, znB, zt_w2, nullptr, nullptr, bufP, 65536, 512, 512, 128, 0, 0, 0);
  launch_gemm(stream, 1, bufP, zt_w3, nullptr, nullptr, nullptr, Zw, nullptr, 65536, 128, 128, 512, 0, 0, 0);

  // =================== pair-bias attention on S ===================
  ln(Sw, 43, 44, snB, 256, 384, 0);
  ln(Zw, 50, 51, znB, 65536, 128, 0);
  launch_gemm(stream, 0, znB, apb_wbz, nullptr, nullptr, nullptr, biasF, nullptr, 65536, 16, 12, 128, 1, 0, 0);
  launch_gemm(stream, 2, snB, apb_wq, nullptr, nullptr, (const float*)d_in[46], nullptr, bufA, 256, 384, 384, 384, 4, 8192, 0);
  launch_gemm(stream, 2, snB, apb_wk, nullptr, nullptr, nullptr, nullptr, bufB, 256, 384, 384, 384, 4, 8192, 0);
  launch_gemm(stream, 2, snB, apb_wv, nullptr, nullptr, nullptr, nullptr, bufV, 256, 384, 384, 384, 5, 8192, 0);
  launch_gemm(stream, 6, snB, apb_wg, nullptr, nullptr, nullptr, nullptr, bufG, 256, 384, 384, 384, 4, 8192, 0);
  attn_softmax_kernel<<<12, 128, 0, stream>>>(bufA, bufB, biasF, bufP, 0, SCALE_ATT);
  attn_pv_kernel<<<12, 256, 0, stream>>>(bufP, bufV, bufG, goB, 0, 384);
  launch_gemm(stream, 1, goB, apb_wo, nullptr, nullptr, nullptr, Sw, nullptr, 256, 384, 384, 384, 0, 0, 0);

  // =================== S transition ===================
  ln(Sw, 54, 55, snB, 256, 384, 0);
  launch_gemm(stream, 4, snB, st_w1, snB, st_w2, nullptr, nullptr, bufP, 256, 1536, 1536, 384, 0, 0, 0);
  launch_gemm(stream, 1, bufP, st_w3, nullptr, nullptr, nullptr, Sw, nullptr, 256, 384, 384, 1536, 0, 0, 0);
}
